// AttentionBlock_64957085385124
// MI455X (gfx1250) — matrix operand order repaired and checked by reading, not compiled, not run on hardware
//
#include <hip/hip_runtime.h>

typedef __bf16 bf16;
typedef __attribute__((ext_vector_type(16))) __bf16 v16bf;
typedef __attribute__((ext_vector_type(8)))  __bf16 bf16x8;   // 16 bytes
typedef __attribute__((ext_vector_type(8)))  float  v8f;

union Frag { v16bf v; bf16x8 h[2]; };

__device__ __forceinline__ v8f wmma_bf16(v16bf a, v16bf b, v8f c) {
  // D = A(16x32 bf16) * B(32x16 bf16) + C(16x16 f32)
  return __builtin_amdgcn_wmma_f32_16x16x32_bf16(
      /*neg_a=*/false, a, /*neg_b=*/false, b,
      /*c_mod=*/(short)0, c, /*reuse_a=*/false, /*reuse_b=*/false);
}

// ---------------------------------------------------------------- constants
#define Bb 8
#define Cc 512
#define NH 4
#define HD 128
#define NN 1024          // H*W
#define GROUPS 32
#define CPG 16           // channels per group

// ---------------------------------------------------------------- fp32->bf16
__global__ void cvt_kernel(const float* __restrict__ src, bf16* __restrict__ dst, int n) {
  int i = blockIdx.x * 256 + threadIdx.x;
  if (i < n) dst[i] = (bf16)src[i];
}

// ---------------------------------------------------------------- GroupNorm
// one block per (b, group); writes xn_t[b][n][c] (bf16, c contiguous)
__global__ __launch_bounds__(256)
void gn_kernel(const float* __restrict__ x, const float* __restrict__ gamma,
               const float* __restrict__ beta, bf16* __restrict__ xn_t) {
  const int bg = blockIdx.x;
  const int b = bg >> 5, g = bg & 31;
  const int cbase = g * CPG;
  const float* xb = x + ((size_t)b * Cc + cbase) * NN;

  float s1 = 0.f, s2 = 0.f;
  for (int i = threadIdx.x; i < CPG * NN; i += 256) {
    float t = xb[i];
    s1 += t; s2 += t * t;
  }
  for (int m = 16; m >= 1; m >>= 1) {
    s1 += __shfl_xor(s1, m, 32);
    s2 += __shfl_xor(s2, m, 32);
  }
  __shared__ float red[2][8];
  const int wave = threadIdx.x >> 5, lane = threadIdx.x & 31;
  if (lane == 0) { red[0][wave] = s1; red[1][wave] = s2; }
  __syncthreads();
  s1 = red[0][lane & 7];
  s2 = red[1][lane & 7];
  for (int m = 4; m >= 1; m >>= 1) {
    s1 += __shfl_xor(s1, m, 32);
    s2 += __shfl_xor(s2, m, 32);
  }
  const float invN = 1.0f / (float)(CPG * NN);
  const float mean = s1 * invN;
  const float var  = s2 * invN - mean * mean;
  const float rstd = rsqrtf(var + 1e-5f);

  for (int i = threadIdx.x; i < CPG * NN; i += 256) {
    const int cc = i >> 10, n = i & (NN - 1);
    const int c = cbase + cc;
    const float t = (xb[i] - mean) * rstd * gamma[c] + beta[c];
    xn_t[((size_t)b * NN + n) * Cc + c] = (bf16)t;
  }
}

// ---------------------------------------------------------------- QKV GEMM
// qkv[o,n] = sum_c W[o,c] * xn[c,n]; per wave 32x32 tile (2x2 WMMA).
// q,k stored transposed [b][h][n][d]; v stored [b][h][d][m].
__global__ __launch_bounds__(256)
void qkv_gemm_kernel(const bf16* __restrict__ W, const bf16* __restrict__ xn_t,
                     bf16* __restrict__ q_t, bf16* __restrict__ k_t,
                     bf16* __restrict__ vmat) {
  const int b = blockIdx.z;
  const int wave = threadIdx.x >> 5, lane = threadIdx.x & 31;
  const int lm = lane & 15, lh = lane >> 4;
  const int m0 = blockIdx.y * 64 + (wave >> 2) * 32;
  const int n0 = blockIdx.x * 128 + (wave & 3) * 32;

  const bf16* Bm = xn_t + (size_t)b * NN * Cc;
  const bf16* ar0 = W  + (size_t)(m0 + lm) * Cc;
  const bf16* ar1 = W  + (size_t)(m0 + 16 + lm) * Cc;
  const bf16* br0 = Bm + (size_t)(n0 + lm) * Cc;
  const bf16* br1 = Bm + (size_t)(n0 + 16 + lm) * Cc;

  v8f acc[2][2] = {};
  for (int k0 = 0; k0 < Cc; k0 += 32) {
    Frag a0, a1, b0, b1;
    const int kh = k0 + lh * 8;
    a0.h[0] = *(const bf16x8*)(ar0 + kh);  a0.h[1] = *(const bf16x8*)(ar0 + kh + 16);
    a1.h[0] = *(const bf16x8*)(ar1 + kh);  a1.h[1] = *(const bf16x8*)(ar1 + kh + 16);
    const int kb = k0 + lh * 8;
    b0.h[0] = *(const bf16x8*)(br0 + kb);  b0.h[1] = *(const bf16x8*)(br0 + kb + 16);
    b1.h[0] = *(const bf16x8*)(br1 + kb);  b1.h[1] = *(const bf16x8*)(br1 + kb + 16);
    acc[0][0] = wmma_bf16(a0.v, b0.v, acc[0][0]);
    acc[0][1] = wmma_bf16(a0.v, b1.v, acc[0][1]);
    acc[1][0] = wmma_bf16(a1.v, b0.v, acc[1][0]);
    acc[1][1] = wmma_bf16(a1.v, b1.v, acc[1][1]);
  }
  for (int i = 0; i < 2; ++i)
    for (int j = 0; j < 2; ++j)
      for (int r = 0; r < 8; ++r) {
        const int o = m0 + 16 * i + r + 8 * lh;
        const int n = n0 + 16 * j + lm;
        const bf16 hv = (bf16)acc[i][j][r];
        const int sec = o >> 9;        // 0:q 1:k 2:v
        const int oc  = o & (Cc - 1);
        const int hh  = oc >> 7, d = oc & (HD - 1);
        const size_t bh = (size_t)b * NH + hh;
        if (sec == 0)      q_t[(bh * NN + n) * HD + d] = hv;
        else if (sec == 1) k_t[(bh * NN + n) * HD + d] = hv;
        else               vmat[(bh * HD + d) * NN + n] = hv;
      }
}

// ---------------------------------------------------------------- attention
// one wave per (b, h, 16-query tile); online softmax over m in chunks of 32.
__global__ __launch_bounds__(256)
void attn_kernel(const bf16* __restrict__ q_t, const bf16* __restrict__ k_t,
                 const bf16* __restrict__ vmat, bf16* __restrict__ attn_t) {
  __shared__ bf16 plds[8][16 * 32];   // per-wave P tile (row-major 16x32)
  const int wave = threadIdx.x >> 5, lane = threadIdx.x & 31;
  const int lm = lane & 15, lh = lane >> 4;
  const int wg = blockIdx.x * 8 + wave;          // 0..2047
  const int n0 = (wg & 63) * 16;
  const int hh = (wg >> 6) & 3;
  const int b  = wg >> 8;
  const size_t bh = (size_t)b * NH + hh;
  const bf16* qb = q_t + bh * NN * HD;
  const bf16* kb = k_t + bh * NN * HD;
  const bf16* vb = vmat + bh * HD * NN;
  const float scale = 0.04419417382415922f;      // 512^-0.5

  Frag qf[4];
  {
    const bf16* qrow = qb + (size_t)(n0 + lm) * HD;
    for (int kk = 0; kk < 4; ++kk) {
      const int kh = kk * 32 + lh * 8;
      qf[kk].h[0] = *(const bf16x8*)(qrow + kh);
      qf[kk].h[1] = *(const bf16x8*)(qrow + kh + 16);
    }
  }

  v8f o[8] = {};
  float mi[8], li[8];
  for (int r = 0; r < 8; ++r) { mi[r] = -3.0e38f; li[r] = 0.f; }
  bf16* pl = &plds[wave][0];

  for (int m0 = 0; m0 < NN; m0 += 32) {
    v8f s0 = {}, s1 = {};
    for (int kk = 0; kk < 4; ++kk) {
      const int kc = kk * 32 + lh * 8;
      const bf16* kr0 = kb + (size_t)(m0 + lm) * HD + kc;
      const bf16* kr1 = kb + (size_t)(m0 + 16 + lm) * HD + kc;
      Frag b0, b1;
      b0.h[0] = *(const bf16x8*)(kr0); b0.h[1] = *(const bf16x8*)(kr0 + 16);
      b1.h[0] = *(const bf16x8*)(kr1); b1.h[1] = *(const bf16x8*)(kr1 + 16);
      s0 = wmma_bf16(qf[kk].v, b0.v, s0);
      s1 = wmma_bf16(qf[kk].v, b1.v, s1);
    }
    for (int r = 0; r < 8; ++r) {
      float a0 = s0[r] * scale, a1 = s1[r] * scale;
      float mx = fmaxf(a0, a1);
      for (int msk = 8; msk >= 1; msk >>= 1) mx = fmaxf(mx, __shfl_xor(mx, msk, 32));
      const float mnew = fmaxf(mi[r], mx);
      const float p0 = __expf(a0 - mnew), p1 = __expf(a1 - mnew);
      float rs = p0 + p1;
      for (int msk = 8; msk >= 1; msk >>= 1) rs += __shfl_xor(rs, msk, 32);
      const float corr = __expf(mi[r] - mnew);
      li[r] = li[r] * corr + rs;
      mi[r] = mnew;
      for (int t = 0; t < 8; ++t) o[t][r] *= corr;
      const int row = r + 8 * lh;
      pl[row * 32 + lm]      = (bf16)p0;
      pl[row * 32 + 16 + lm] = (bf16)p1;
    }
    // re-fragment P (C layout -> A layout) through LDS; same-wave, waits auto
    Frag pa;
    {
      const bf16* pr = pl + lm * 32 + lh * 8;
      pa.h[0] = *(const bf16x8*)(pr);
      pa.h[1] = *(const bf16x8*)(pr + 16);
    }
    for (int t = 0; t < 8; ++t) {
      const bf16* vr = vb + (size_t)(t * 16 + lm) * NN + m0 + lh * 8;
      Frag vf;
      vf.h[0] = *(const bf16x8*)(vr); vf.h[1] = *(const bf16x8*)(vr + 16);
      o[t] = wmma_bf16(pa.v, vf.v, o[t]);
    }
  }

  for (int r = 0; r < 8; ++r) {
    const float inv = 1.0f / li[r];
    const int n = n0 + r + 8 * lh;
    for (int t = 0; t < 8; ++t) {
      const int c = hh * HD + t * 16 + lm;
      attn_t[((size_t)b * NN + n) * Cc + c] = (bf16)(o[t][r] * inv);
    }
  }
}

// ---------------------------------------------------------------- proj GEMM
__global__ __launch_bounds__(256)
void proj_gemm_kernel(const bf16* __restrict__ W, const bf16* __restrict__ attn_t,
                      const float* __restrict__ b_proj, const float* __restrict__ x,
                      float* __restrict__ out) {
  const int b = blockIdx.z;
  const int wave = threadIdx.x >> 5, lane = threadIdx.x & 31;
  const int lm = lane & 15, lh = lane >> 4;
  const int m0 = blockIdx.y * 64 + (wave >> 2) * 32;
  const int n0 = blockIdx.x * 128 + (wave & 3) * 32;

  const bf16* Bm = attn_t + (size_t)b * NN * Cc;
  const bf16* ar0 = W  + (size_t)(m0 + lm) * Cc;
  const bf16* ar1 = W  + (size_t)(m0 + 16 + lm) * Cc;
  const bf16* br0 = Bm + (size_t)(n0 + lm) * Cc;
  const bf16* br1 = Bm + (size_t)(n0 + 16 + lm) * Cc;

  v8f acc[2][2] = {};
  for (int k0 = 0; k0 < Cc; k0 += 32) {
    Frag a0, a1, b0, b1;
    const int kh = k0 + lh * 8;
    a0.h[0] = *(const bf16x8*)(ar0 + kh);  a0.h[1] = *(const bf16x8*)(ar0 + kh + 16);
    a1.h[0] = *(const bf16x8*)(ar1 + kh);  a1.h[1] = *(const bf16x8*)(ar1 + kh + 16);
    const int kb = k0 + lh * 8;
    b0.h[0] = *(const bf16x8*)(br0 + kb);  b0.h[1] = *(const bf16x8*)(br0 + kb + 16);
    b1.h[0] = *(const bf16x8*)(br1 + kb);  b1.h[1] = *(const bf16x8*)(br1 + kb + 16);
    acc[0][0] = wmma_bf16(a0.v, b0.v, acc[0][0]);
    acc[0][1] = wmma_bf16(a0.v, b1.v, acc[0][1]);
    acc[1][0] = wmma_bf16(a1.v, b0.v, acc[1][0]);
    acc[1][1] = wmma_bf16(a1.v, b1.v, acc[1][1]);
  }
  for (int i = 0; i < 2; ++i)
    for (int j = 0; j < 2; ++j)
      for (int r = 0; r < 8; ++r) {
        const int oo = m0 + 16 * i + r + 8 * lh;
        const int n  = n0 + 16 * j + lm;
        const size_t idx = ((size_t)b * Cc + oo) * NN + n;
        out[idx] = acc[i][j][r] + b_proj[oo] + x[idx];
      }
}

// ---------------------------------------------------------------- launcher
extern "C" void kernel_launch(void* const* d_in, const int* in_sizes, int n_in,
                              void* d_out, int out_size, void* d_ws, size_t ws_size,
                              hipStream_t stream) {
  const float* x      = (const float*)d_in[0];
  const float* gamma  = (const float*)d_in[1];
  const float* beta   = (const float*)d_in[2];
  const float* w_qkv  = (const float*)d_in[3];
  const float* w_proj = (const float*)d_in[4];
  const float* b_proj = (const float*)d_in[5];
  float* out = (float*)d_out;

  char* ws = (char*)d_ws;
  size_t off = 0;
  auto alloc = [&](size_t bytes) {
    char* p = ws + off;
    off += (bytes + 255) & ~(size_t)255;
    return p;
  };
  bf16* w_qkv_b  = (bf16*)alloc((size_t)3 * Cc * Cc * 2);
  bf16* w_proj_b = (bf16*)alloc((size_t)Cc * Cc * 2);
  bf16* xn_t     = (bf16*)alloc((size_t)Bb * NN * Cc * 2);
  bf16* q_t      = (bf16*)alloc((size_t)Bb * NH * NN * HD * 2);
  bf16* k_t      = (bf16*)alloc((size_t)Bb * NH * NN * HD * 2);
  bf16* vmat     = (bf16*)alloc((size_t)Bb * Cc * NN * 2);
  bf16* attn_t   = (bf16*)alloc((size_t)Bb * NN * Cc * 2);
  (void)ws_size; (void)in_sizes; (void)n_in; (void)out_size;

  cvt_kernel<<<(3 * Cc * Cc + 255) / 256, 256, 0, stream>>>(w_qkv, w_qkv_b, 3 * Cc * Cc);
  cvt_kernel<<<(Cc * Cc + 255) / 256, 256, 0, stream>>>(w_proj, w_proj_b, Cc * Cc);
  gn_kernel<<<Bb * GROUPS, 256, 0, stream>>>(x, gamma, beta, xn_t);
  qkv_gemm_kernel<<<dim3(NN / 128, (3 * Cc) / 64, Bb), 256, 0, stream>>>(
      w_qkv_b, xn_t, q_t, k_t, vmat);
  attn_kernel<<<(Bb * NH * (NN / 16)) / 8, 256, 0, stream>>>(q_t, k_t, vmat, attn_t);
  proj_gemm_kernel<<<dim3(NN / 128, Cc / 64, Bb), 256, 0, stream>>>(
      w_proj_b, attn_t, b_proj, x, out);
}
